// Layer14_39290360824167
// MI455X (gfx1250) — matrix operand order repaired and checked by reading, not compiled, not run on hardware
//
#include <hip/hip_runtime.h>

typedef __attribute__((ext_vector_type(16))) _Float16 v16h;
typedef __attribute__((ext_vector_type(8)))  _Float16 v8h;
typedef __attribute__((ext_vector_type(8)))  float    v8f;
typedef __attribute__((ext_vector_type(4)))  float    v4f;

#define SQ    2048
#define HH    1024
#define NHEAD 16
#define HDIM  64
#define BATCH 4

static __device__ __forceinline__ v8f wmma_f16(v16h a, v16h b, v8f c) {
    // D = A(16x32 f16) x B(32x16 f16) + C(16x16 f32)
    return __builtin_amdgcn_wmma_f32_16x16x32_f16(
        /*neg_a=*/false, a, /*neg_b=*/false, b,
        /*c_mod=*/(short)0, c, /*reuse_a=*/false, /*reuse_b=*/false);
}

static __device__ __forceinline__ v8f vzero8() {
    v8f z;
    #pragma unroll
    for (int i = 0; i < 8; ++i) z[i] = 0.0f;
    return z;
}

#define ASYNC_B128(ldsoff, gaddr)                                               \
    asm volatile("global_load_async_to_lds_b128 %0, %1, off"                    \
                 :: "v"(ldsoff), "v"(gaddr) : "memory")

// Zero-instruction sink: forces all listed 128-bit operands to be materialized
// here, so the scheduler batches their loads and emits ONE wait.
#define SINK8(a, b, c, d, e, f, g, h)                                           \
    asm volatile("" :: "v"(a), "v"(b), "v"(c), "v"(d),                          \
                       "v"(e), "v"(f), "v"(g), "v"(h))

// ---- v_permlane16_b32 butterfly reductions within 16-lane rows (no LDS) ----
static __device__ __forceinline__ float pl16(float v, unsigned s0, unsigned s1) {
    int x = __float_as_int(v);
    int r = __builtin_amdgcn_permlane16(x, x, s0, s1, false, false);
    return __int_as_float(r);
}
static __device__ __forceinline__ float rowmax16(float v) {
    v = fmaxf(v, pl16(v, 0x67452301u, 0xEFCDAB89u));   // xor 1
    v = fmaxf(v, pl16(v, 0x54761032u, 0xDCFE98BAu));   // xor 2
    v = fmaxf(v, pl16(v, 0x32107654u, 0xBA98FEDCu));   // xor 4
    v = fmaxf(v, pl16(v, 0xFEDCBA98u, 0x76543210u));   // xor 8
    return v;
}
static __device__ __forceinline__ float rowsum16(float v) {
    v = v + pl16(v, 0x67452301u, 0xEFCDAB89u);
    v = v + pl16(v, 0x54761032u, 0xDCFE98BAu);
    v = v + pl16(v, 0x32107654u, 0xBA98FEDCu);
    v = v + pl16(v, 0xFEDCBA98u, 0x76543210u);
    return v;
}

// ---------------------------------------------------------------------------
// Kernel 0: f32 -> f16 conversion (memory-bound, ~3us total at 23.3 TB/s).
// ---------------------------------------------------------------------------
__global__ void cvt_f32_f16(const float* __restrict__ src,
                            _Float16* __restrict__ dst, int n8) {
    const int i = blockIdx.x * blockDim.x + threadIdx.x;
    if (i < n8) {
        v4f a = *(const v4f*)(src + (size_t)i * 8);
        v4f b = *(const v4f*)(src + (size_t)i * 8 + 4);
        v8h o;
        #pragma unroll
        for (int e = 0; e < 4; ++e) {
            o[e]     = (_Float16)a[e];
            o[4 + e] = (_Float16)b[e];
        }
        *(v8h*)(dst + (size_t)i * 8) = o;
    }
}

// ---------------------------------------------------------------------------
// Kernel A: QKV projection GEMM, pure f16.
// Block = 8 waves computes 256(M) x 64(N); each wave owns 32 M-rows (two A
// tiles) and runs 8 WMMAs per 32-wide k-step against 4 shared B tiles.
// Weight tile async-staged in LDS, double-buffered, unrolled schedule.
// Q,K stored [B,NH,S,HD]; V stored transposed [B,NH,HD,S].
// ---------------------------------------------------------------------------
#define WPAD 40                       // 32 + 8 halves: 80B stride, conflict-free

__global__ void qkv_gemm(const _Float16* __restrict__ Xh,
                         const _Float16* __restrict__ Wqh,
                         const _Float16* __restrict__ Wkh,
                         const _Float16* __restrict__ Wvh,
                         const float* __restrict__ bq,
                         const float* __restrict__ bk,
                         const float* __restrict__ bv,
                         _Float16* __restrict__ qo,
                         _Float16* __restrict__ ko,
                         _Float16* __restrict__ vo) {
    __shared__ __align__(16) _Float16 wbuf[2][64 * WPAD];

    const int tid  = threadIdx.x;
    const int lane = tid & 31;
    const int wave = tid >> 5;

    const int gb  = blockIdx.x;
    const int o   = gb / (32 * 16);          // 0=Q 1=K 2=V
    const int rem = gb % (32 * 16);
    const int m0  = (rem / 16) * 256 + wave * 32;
    const int n0  = (rem % 16) * 64;

    const _Float16* W    = (o == 0) ? Wqh : ((o == 1) ? Wkh : Wvh);
    const float*    bias = (o == 0) ? bq  : ((o == 1) ? bk  : bv);

    const int half = (lane >= 16) ? 1 : 0;
    const int l15  = lane & 15;

    // loader: each thread moves one 16B segment per k-step
    const int lrow = tid >> 2;
    const int lseg = (tid & 3) * 8;
    const _Float16* gsrc = W + (size_t)(n0 + lrow) * HH + lseg;
    const unsigned ldsoff0 = (unsigned)(size_t)(void*)&wbuf[0][lrow * WPAD + lseg];
    const unsigned ldsoff1 = (unsigned)(size_t)(void*)&wbuf[1][lrow * WPAD + lseg];

    v8f acc[2][4];
    #pragma unroll
    for (int r = 0; r < 2; ++r)
        #pragma unroll
        for (int t = 0; t < 4; ++t) acc[r][t] = vzero8();

    const _Float16* arow0 = Xh + (size_t)(m0 + l15) * HH;
    const _Float16* arow1 = arow0 + (size_t)16 * HH;

    auto body = [&](int bufi, int kb, bool prefetch) {
        if (prefetch) {
            ASYNC_B128((bufi ? ldsoff0 : ldsoff1),
                       (unsigned long long)(size_t)(const void*)(gsrc + kb + 32));
            asm volatile("s_wait_asynccnt 0x1" ::: "memory");
        } else {
            asm volatile("s_wait_asynccnt 0x0" ::: "memory");
        }
        __syncthreads();

        // two A tiles (rows m0..m0+15 and m0+16..m0+31)
        const int ksel = kb + (half ? 8 : 0);
        v8h alo0 = *(const v8h*)(arow0 + ksel);
        v8h ahi0 = *(const v8h*)(arow0 + ksel + 16);
        v8h alo1 = *(const v8h*)(arow1 + ksel);
        v8h ahi1 = *(const v8h*)(arow1 + ksel + 16);

        // 4 shared B tiles from LDS
        v8h wb[4][2];
        const _Float16* wbase = &wbuf[bufi][0];
        #pragma unroll
        for (int t = 0; t < 4; ++t) {
            const _Float16* wr = wbase + (t * 16 + l15) * WPAD + (half ? 8 : 0);
            wb[t][0] = *(const v8h*)(wr);
            wb[t][1] = *(const v8h*)(wr + 16);
        }
        SINK8(wb[0][0], wb[0][1], wb[1][0], wb[1][1],
              wb[2][0], wb[2][1], wb[3][0], wb[3][1]);
        SINK8(alo0, ahi0, alo1, ahi1, alo0, ahi0, alo1, ahi1);

        v16h A0, A1;
        #pragma unroll
        for (int e = 0; e < 8; ++e) {
            A0[e] = alo0[e]; A0[8 + e] = ahi0[e];
            A1[e] = alo1[e]; A1[8 + e] = ahi1[e];
        }
        #pragma unroll
        for (int t = 0; t < 4; ++t) {
            v16h Bt;
            #pragma unroll
            for (int e = 0; e < 8; ++e) { Bt[e] = wb[t][0][e]; Bt[8 + e] = wb[t][1][e]; }
            acc[0][t] = wmma_f16(A0, Bt, acc[0][t]);
            acc[1][t] = wmma_f16(A1, Bt, acc[1][t]);
        }
        __syncthreads();
    };

    ASYNC_B128(ldsoff0, (unsigned long long)(size_t)(const void*)gsrc);
    for (int kb = 0; kb < HH - 64; kb += 64) {
        body(0, kb, true);
        body(1, kb + 32, true);
    }
    body(0, HH - 64, true);
    body(1, HH - 32, false);

    // bias + store (D layout: lane = column n, vgpr slot = row)
    #pragma unroll
    for (int t = 0; t < 4; ++t) {
        const int n  = n0 + t * 16 + l15;
        const float bv_ = bias[n];
        const int h = n >> 6;
        const int d = n & 63;
        #pragma unroll
        for (int r = 0; r < 2; ++r) {
            #pragma unroll
            for (int i = 0; i < 8; ++i) {
                const int m = m0 + r * 16 + i + half * 8;
                const int b = m >> 11;
                const int s = m & 2047;
                const float val = acc[r][t][i] + bv_;
                if (o == 2) {
                    vo[((size_t)(b * NHEAD + h) * HDIM + d) * SQ + s] = (_Float16)val;
                } else {
                    _Float16* dst = (o == 0) ? qo : ko;
                    dst[((size_t)(b * NHEAD + h) * SQ + s) * HDIM + d] = (_Float16)val;
                }
            }
        }
    }
}

// ---------------------------------------------------------------------------
// Kernel B: cooperative flash attention.
// Block = 8 waves, all on the SAME (b,h), query tiles q0 = qblk*128 + wave*16.
// K tile (32x64) and V tile (64x32, from V^T) async-staged in LDS, shared by
// all 8 waves, double-buffered, unrolled schedule. Softmax reductions use
// v_permlane16_b32 butterflies (no LDS, no counter waits).
// ---------------------------------------------------------------------------
#define KPAD 72   // 64 + 8 halves: 144B stride = 36 dwords, conflict-free
#define VPAD 40   // 32 + 8 halves: 80B stride  = 20 dwords, conflict-free

__global__ void attn_kernel(const _Float16* __restrict__ qb,
                            const _Float16* __restrict__ kbuf,
                            const _Float16* __restrict__ vt,
                            const float* __restrict__ mask,
                            float* __restrict__ out) {
    __shared__ __align__(16) _Float16 ktile[2][32 * KPAD];   // 2 x 4608 B
    __shared__ __align__(16) _Float16 vtile[2][64 * VPAD];   // 2 x 5120 B
    __shared__ __align__(16) _Float16 pbuf[8][16 * 32];      // per-wave P transpose

    const int tid  = threadIdx.x;
    const int lane = tid & 31;
    const int wave = tid >> 5;

    const int gb = blockIdx.x;
    const int b  = gb / (NHEAD * (SQ / 128));
    const int h  = (gb / (SQ / 128)) % NHEAD;
    const int q0 = (gb % (SQ / 128)) * 128 + wave * 16;

    const int half = (lane >= 16) ? 1 : 0;
    const int l15  = lane & 15;

    const _Float16* kbase = kbuf + (size_t)(b * NHEAD + h) * SQ * HDIM;
    const _Float16* vbase = vt   + (size_t)(b * NHEAD + h) * HDIM * SQ;
    const float*    mrow  = mask + (size_t)b * SQ;

    // loader mappings (one 16B async segment each for K and V)
    const int krow_l = tid >> 3;
    const int kseg_l = (tid & 7) * 8;
    const _Float16* ksrc = kbase + (size_t)krow_l * HDIM + kseg_l;
    const unsigned koff0 = (unsigned)(size_t)(void*)&ktile[0][krow_l * KPAD + kseg_l];
    const unsigned koff1 = (unsigned)(size_t)(void*)&ktile[1][krow_l * KPAD + kseg_l];

    const int vrow_l = tid >> 2;
    const int vseg_l = (tid & 3) * 8;
    const _Float16* vsrc = vbase + (size_t)vrow_l * SQ + vseg_l;
    const unsigned voff0 = (unsigned)(size_t)(void*)&vtile[0][vrow_l * VPAD + vseg_l];
    const unsigned voff1 = (unsigned)(size_t)(void*)&vtile[1][vrow_l * VPAD + vseg_l];

    // Q tile into two A-regs (16x32 each), folded 1/sqrt(HD)
    const _Float16* qrow = qb + ((size_t)(b * NHEAD + h) * SQ + q0 + l15) * HDIM;
    v16h a0, a1;
    {
        const int dsel = half ? 8 : 0;
        v8h lo  = *(const v8h*)(qrow + dsel);
        v8h hi  = *(const v8h*)(qrow + dsel + 16);
        v8h lo2 = *(const v8h*)(qrow + 32 + dsel);
        v8h hi2 = *(const v8h*)(qrow + 32 + dsel + 16);
        const _Float16 sc = (_Float16)0.125f;   // 1/sqrt(64), exact in f16
        #pragma unroll
        for (int e = 0; e < 8; ++e) {
            a0[e]     = lo[e]  * sc;
            a0[8 + e] = hi[e]  * sc;
            a1[e]     = lo2[e] * sc;
            a1[8 + e] = hi2[e] * sc;
        }
    }

    float mi[8], li[8];
    v8f ctx[4];
    #pragma unroll
    for (int i = 0; i < 8; ++i) { mi[i] = -1e30f; li[i] = 0.0f; }
    #pragma unroll
    for (int t = 0; t < 4; ++t) ctx[t] = vzero8();

    auto body = [&](int bufi, int k0, bool prefetch) {
        if (prefetch) {
            ASYNC_B128((bufi ? koff0 : koff1),
                (unsigned long long)(size_t)(const void*)(ksrc + (size_t)(k0 + 32) * HDIM));
            ASYNC_B128((bufi ? voff0 : voff1),
                (unsigned long long)(size_t)(const void*)(vsrc + k0 + 32));
            asm volatile("s_wait_asynccnt 0x2" ::: "memory");
        } else {
            asm volatile("s_wait_asynccnt 0x0" ::: "memory");
        }
        __syncthreads();

        // K B-operands from LDS (8x ds_load_b128), batched, then score WMMAs
        const _Float16* kt = bufi ? &ktile[1][0] : &ktile[0][0];
        const _Float16* vl = bufi ? &vtile[1][0] : &vtile[0][0];
        v8h kx[2][4];
        float mval[2];
        #pragma unroll
        for (int t = 0; t < 2; ++t) {
            const _Float16* kr = kt + (t * 16 + l15) * KPAD + (half ? 8 : 0);
            kx[t][0] = *(const v8h*)(kr);
            kx[t][1] = *(const v8h*)(kr + 16);
            kx[t][2] = *(const v8h*)(kr + 32);
            kx[t][3] = *(const v8h*)(kr + 48);
            mval[t]  = mrow[k0 + t * 16 + l15];
        }
        SINK8(kx[0][0], kx[0][1], kx[0][2], kx[0][3],
              kx[1][0], kx[1][1], kx[1][2], kx[1][3]);

        v8f s[2];
        #pragma unroll
        for (int t = 0; t < 2; ++t) {
            v16h b0, b1;
            #pragma unroll
            for (int e = 0; e < 8; ++e) {
                b0[e] = kx[t][0][e]; b0[8 + e] = kx[t][1][e];
                b1[e] = kx[t][2][e]; b1[8 + e] = kx[t][3][e];
            }
            v8f z = vzero8();
            z = wmma_f16(a0, b0, z);
            z = wmma_f16(a1, b1, z);
            #pragma unroll
            for (int i = 0; i < 8; ++i) z[i] += mval[t];
            s[t] = z;
        }

        // online softmax; rows live across the 16-lane half-groups.
        // Reductions are v_permlane16 butterflies: pure VALU, no waits.
        float pa0[8], pa1[8];
        #pragma unroll
        for (int i = 0; i < 8; ++i) {
            const float mb   = rowmax16(fmaxf(s[0][i], s[1][i]));
            const float mn   = fmaxf(mi[i], mb);
            const float corr = __expf(mi[i] - mn);
            const float p0 = __expf(s[0][i] - mn);
            const float p1 = __expf(s[1][i] - mn);
            const float rs = rowsum16(p0 + p1);
            li[i] = li[i] * corr + rs;
            mi[i] = mn;
            pa0[i] = p0; pa1[i] = p1;
            #pragma unroll
            for (int t = 0; t < 4; ++t) ctx[t][i] *= corr;
        }

        // V B-operands issued now; the dscnt-0 wait below covers them too
        v8h vx0[4], vx1[4];
        #pragma unroll
        for (int t = 0; t < 4; ++t) {
            const _Float16* vr = vl + (t * 16 + l15) * VPAD + (half ? 8 : 0);
            vx0[t] = *(const v8h*)(vr);
            vx1[t] = *(const v8h*)(vr + 16);
        }

        // D-layout P -> A-layout P via per-wave LDS scratch
        _Float16* pb = &pbuf[wave][0];
        #pragma unroll
        for (int i = 0; i < 8; ++i) {
            const int row = i + half * 8;
            pb[row * 32 + l15]      = (_Float16)pa0[i];
            pb[row * 32 + 16 + l15] = (_Float16)pa1[i];
        }
        asm volatile("s_wait_dscnt 0x0" ::: "memory");
        v16h pA;
        {
            const int c0 = half ? 8 : 0;
            v8h plo = *(const v8h*)(pb + l15 * 32 + c0);
            v8h phi = *(const v8h*)(pb + l15 * 32 + c0 + 16);
            #pragma unroll
            for (int e = 0; e < 8; ++e) { pA[e] = plo[e]; pA[8 + e] = phi[e]; }
        }

        // ctx += P(16x32) x V(32x16 per d-tile)
        #pragma unroll
        for (int t = 0; t < 4; ++t) {
            v16h vB;
            #pragma unroll
            for (int e = 0; e < 8; ++e) { vB[e] = vx0[t][e]; vB[8 + e] = vx1[t][e]; }
            ctx[t] = wmma_f16(pA, vB, ctx[t]);
        }
        __syncthreads();
    };

    ASYNC_B128(koff0, (unsigned long long)(size_t)(const void*)ksrc);
    ASYNC_B128(voff0, (unsigned long long)(size_t)(const void*)vsrc);
    for (int k0 = 0; k0 < SQ - 64; k0 += 64) {
        body(0, k0, true);
        body(1, k0 + 32, true);
    }
    body(0, SQ - 64, true);
    body(1, SQ - 32, false);

    // normalize and write out [B,S,H] f32
    #pragma unroll
    for (int i = 0; i < 8; ++i) {
        const float inv = 1.0f / li[i];
        const int row = q0 + i + half * 8;
        float* orow = out + ((size_t)(b * SQ + row)) * HH + h * HDIM;
        #pragma unroll
        for (int t = 0; t < 4; ++t) {
            orow[t * 16 + l15] = ctx[t][i] * inv;
        }
    }
}

extern "C" void kernel_launch(void* const* d_in, const int* in_sizes, int n_in,
                              void* d_out, int out_size, void* d_ws, size_t ws_size,
                              hipStream_t stream) {
    const float* X    = (const float*)d_in[0];   // [4,2048,1024]
    const float* mask = (const float*)d_in[1];   // [4,1,1,2048]
    const float* Wq   = (const float*)d_in[2];
    const float* bq   = (const float*)d_in[3];
    const float* Wk   = (const float*)d_in[4];
    const float* bk   = (const float*)d_in[5];
    const float* Wv   = (const float*)d_in[6];
    const float* bv   = (const float*)d_in[7];
    float* out = (float*)d_out;

    const size_t tok_elems = (size_t)BATCH * SQ * HH;   // 8388608
    const size_t w_elems   = (size_t)HH * HH;           // 1048576

    _Float16* qbuf = (_Float16*)d_ws;
    _Float16* kbuf = qbuf + tok_elems;
    _Float16* vbuf = kbuf + tok_elems;
    _Float16* xh   = vbuf + tok_elems;
    _Float16* wqh  = xh   + tok_elems;
    _Float16* wkh  = wqh  + w_elems;
    _Float16* wvh  = wkh  + w_elems;                    // ~70 MB total

    cvt_f32_f16<<<(int)(tok_elems / 8 / 256), 256, 0, stream>>>(X,  xh,  (int)(tok_elems / 8));
    cvt_f32_f16<<<(int)(w_elems   / 8 / 256), 256, 0, stream>>>(Wq, wqh, (int)(w_elems / 8));
    cvt_f32_f16<<<(int)(w_elems   / 8 / 256), 256, 0, stream>>>(Wk, wkh, (int)(w_elems / 8));
    cvt_f32_f16<<<(int)(w_elems   / 8 / 256), 256, 0, stream>>>(Wv, wvh, (int)(w_elems / 8));

    // 3 outputs * (8192/256 M-blocks) * (1024/64 N-strips) = 1536 blocks
    qkv_gemm<<<1536, 256, 0, stream>>>(xh, wqh, wkh, wvh, bq, bk, bv,
                                       qbuf, kbuf, vbuf);

    // 4*16 heads * (2048/128 q-blocks) = 1024 blocks, 8 waves each
    attn_kernel<<<1024, 256, 0, stream>>>(qbuf, kbuf, vbuf, mask, out);
}
